// GPSConv_76166950027415
// MI455X (gfx1250) — matrix operand order repaired and checked by reading, not compiled, not run on hardware
//
#include <hip/hip_runtime.h>
#include <hip/hip_bf16.h>

// ---------------------------------------------------------------------------
// Problem constants (from reference)
// ---------------------------------------------------------------------------
#define GN 50000            // nodes
#define GL 4                // list dim
#define GD 128              // feature dim
#define GE 800000           // edges
#define ROWS (GL * GN)      // 200000 flattened rows
#define DECAY 0.7f
#define LN_EPS 1e-5f

typedef _Float16 v16h __attribute__((ext_vector_type(16)));
typedef _Float16 h8   __attribute__((ext_vector_type(8)));
typedef float    v8f  __attribute__((ext_vector_type(8)));
typedef float    f4   __attribute__((ext_vector_type(4)));

union AFrag { v16h v; h8 h[2]; };

__device__ __forceinline__ v8f zero_v8f() {
    v8f z;
    #pragma unroll
    for (int i = 0; i < 8; ++i) z[i] = 0.0f;
    return z;
}

__device__ __forceinline__ float silu_f(float x) {
    return x / (1.0f + expf(-x));
}

// XOR-butterfly add step via DS_SWIZZLE_B32 (group-of-32 mode):
// offset imm = {xor[4:0]<<10 | or[4:0]<<5 | and[4:0]} ; and=0x1F, or=0.
// XOR masks <= 8 keep exchanges within the 16-lane C-fragment groups.
template <int XORM>
__device__ __forceinline__ float swz_xor_add(float v) {
    int o = __builtin_amdgcn_ds_swizzle(__float_as_int(v),
                                        (XORM << 10) | 0x1f);
    return v + __int_as_float(o);
}

__device__ __forceinline__ float group16_sum(float v) {
    v = swz_xor_add<1>(v);
    v = swz_xor_add<2>(v);
    v = swz_xor_add<4>(v);
    v = swz_xor_add<8>(v);
    return v;
}

// Async global -> LDS copy of 32 bytes (two B128 ops; IOFFSET applies to both
// the LDS and the global address per CDNA5 ISA 10.x async pseudocode).
__device__ __forceinline__ void async_copy_32B(const void* gsrc, void* lds_dst) {
    unsigned ldsaddr = (unsigned)(size_t)lds_dst;
    asm volatile(
        "global_load_async_to_lds_b128 %0, %1, off\n\t"
        "global_load_async_to_lds_b128 %0, %1, off offset:16"
        :
        : "v"(ldsaddr), "v"(gsrc)
        : "memory");
}

__device__ __forceinline__ void wait_asynccnt0() {
    asm volatile("s_wait_asynccnt 0x0" ::: "memory");
}

// ---------------------------------------------------------------------------
// x (f32) -> f16 elementwise
// ---------------------------------------------------------------------------
__global__ __launch_bounds__(256) void cvt_f32_to_f16_kernel(
    const float* __restrict__ src, _Float16* __restrict__ dst, int n) {
    int i = blockIdx.x * 256 + threadIdx.x;
    if (i < n) dst[i] = (_Float16)src[i];
}

// ---------------------------------------------------------------------------
// Weight swizzle into WMMA B-fragment order.
// Fragment f = ntile*(K/32) + kchunk ; within fragment: lane (0..31), elem e (0..15)
//   n = ntile*16 + (lane&15) ; k = kchunk*32 + (lane>>4)*8 + (e&7) + (e>>3)*16
// out[(f*32 + lane)*16 + e] = W[k, n]   (f32 -> f16)
// Dual-source with split along K (by_k=1) or along N (by_k=0).
// ---------------------------------------------------------------------------
__global__ __launch_bounds__(256) void swizzle_w_kernel(
    const float* __restrict__ W0, int s0,
    const float* __restrict__ W1, int s1,
    int split, int by_k, int K, int N, _Float16* __restrict__ out) {
    int idx = blockIdx.x * 256 + threadIdx.x;
    if (idx >= K * N) return;
    int e    = idx & 15;
    int lane = (idx >> 4) & 31;
    int f    = idx >> 9;
    int KC   = K >> 5;
    int nt   = f / KC;
    int kc   = f - nt * KC;
    int n = nt * 16 + (lane & 15);
    int k = kc * 32 + (lane >> 4) * 8 + (e & 7) + (e >> 3) * 16;
    const float* W = W0; int s = s0; int kk = k; int nn = n;
    if (by_k) {
        if (k >= split) { W = W1; s = s1; kk = k - split; }
    } else {
        if (n >= split) { W = W1; s = s1; nn = n - split; }
    }
    out[idx] = (_Float16)W[(size_t)kk * s + nn];
}

// ---------------------------------------------------------------------------
// COO scatter-add SpMM:  agg[l, r, :] += val * x[l, c, :]
// One block (128 threads over D) per edge, loop over L.
// ---------------------------------------------------------------------------
__global__ __launch_bounds__(128) void spmm_agg_kernel(
    const int* __restrict__ ei, const float* __restrict__ vals,
    const float* __restrict__ x, float* __restrict__ agg, int E) {
    int e = blockIdx.x;
    if (e >= E) return;
    int d = threadIdx.x;
    int r = ei[e];
    int c = ei[E + e];
    float v = vals[e];
    #pragma unroll
    for (int l = 0; l < GL; ++l) {
        size_t ro = ((size_t)l * GN + r) * GD + d;
        size_t co = ((size_t)l * GN + c) * GD + d;
        atomicAdd(&agg[ro], v * x[co]);
    }
}

// ---------------------------------------------------------------------------
// Retention weight propagation: wout[l, r] += DECAY * val * win[l, c]
// ---------------------------------------------------------------------------
__global__ __launch_bounds__(256) void wprop_kernel(
    const int* __restrict__ ei, const float* __restrict__ vals,
    const float* __restrict__ win, float* __restrict__ wout, int E) {
    int e = blockIdx.x * 256 + threadIdx.x;
    if (e >= E) return;
    int r = ei[e];
    int c = ei[E + e];
    float v = vals[e] * DECAY;
    #pragma unroll
    for (int l = 0; l < GL; ++l)
        atomicAdd(&wout[l * GN + r], v * win[l * GN + c]);
}

__global__ __launch_bounds__(256) void waccum_kernel(
    float* __restrict__ wsum, const float* __restrict__ wadd, int n) {
    int i = blockIdx.x * 256 + threadIdx.x;
    if (i < n) wsum[i] += wadd[i];
}

// ---------------------------------------------------------------------------
// Main WMMA GEMM + fused LayerNorm epilogue.
//   C[row, col] = sum_k A[row, k] * B[k, col]   (M = Mtot, N = 128, K = KTOT)
// Block: 256 threads = 8 waves (4 M-tiles x 2 N-tiles); tile 128x128.
// A staged per-32-K-chunk into double-buffered LDS:
//   f16 sources -> GLOBAL_LOAD_ASYNC_TO_LDS_B128 (ASYNCcnt pipelined)
//   f32 sources -> load + inline convert + ds_store
// Software pipeline: issue chunk k+1 staging before computing chunk k.
// Dual A pointers: k < 128 -> A0, k >= 128 -> A1 (KTOT == 256 only).
// Padding rows are clamped (never stored), so no zero-fill branches.
// Epilogue: optional +bias[col], +resid[row,col], *rowscale[row], silu,
// then per-row LayerNorm(g, b); output f16 or f32.
// ---------------------------------------------------------------------------
template <int KTOT, bool A0F16, bool A1F16, bool SILU, bool ROWSCALE,
          bool RESID, bool OUTF32>
__global__ __launch_bounds__(256) void gemm_ln_kernel(
    const void* __restrict__ A0, const void* __restrict__ A1,
    const _Float16* __restrict__ Bw,
    const float* __restrict__ bias,
    const float* __restrict__ rowscale,
    const float* __restrict__ resid,
    const float* __restrict__ ln_g, const float* __restrict__ ln_b,
    void* __restrict__ out, int Mtot) {
    constexpr int KC = KTOT / 32;
    __shared__ __align__(16) _Float16 lda[2][128 * 40];  // padded stride 40
    __shared__ float redS[128][2];
    __shared__ float redS2[128][2];

    const int tid  = threadIdx.x;
    const int wid  = tid >> 5;
    const int lane = tid & 31;
    const int half = lane >> 4;
    const int ln16 = lane & 15;
    const int wave_m = wid & 3;
    const int wave_n = wid >> 2;
    const int blockRow = blockIdx.x * 128;

    // per-thread staging coordinates (fixed across chunks)
    const int srr  = tid >> 1;
    const int skk  = (tid & 1) * 16;
    int srow = blockRow + srr;
    if (srow > Mtot - 1) srow = Mtot - 1;   // clamp padding rows (never stored)

    v8f acc[2][4];
    #pragma unroll
    for (int mi = 0; mi < 2; ++mi)
        #pragma unroll
        for (int ni = 0; ni < 4; ++ni) acc[mi][ni] = zero_v8f();

    // ---- staging issue helper ----
    auto issue_stage = [&](int kc, int bufIdx) {
        _Float16* dst = &lda[bufIdx][srr * 40 + skk];
        const bool useA1 = (KTOT == 256) && (kc >= 4);
        const int kcol = kc * 32 + skk - (useA1 ? 128 : 0);
        const bool srcF16 = useA1 ? A1F16 : A0F16;
        const void* src = useA1 ? A1 : A0;
        if (srcF16) {
            const _Float16* S = (const _Float16*)src + (size_t)srow * GD + kcol;
            async_copy_32B(S, dst);
        } else {
            const float* S = (const float*)src + (size_t)srow * GD + kcol;
            f4 a = *(const f4*)(S);
            f4 b = *(const f4*)(S + 4);
            f4 c = *(const f4*)(S + 8);
            f4 d = *(const f4*)(S + 12);
            h8 lo, hi;
            #pragma unroll
            for (int i = 0; i < 4; ++i) {
                lo[i]     = (_Float16)a[i];
                lo[4 + i] = (_Float16)b[i];
                hi[i]     = (_Float16)c[i];
                hi[4 + i] = (_Float16)d[i];
            }
            *(h8*)(dst)     = lo;
            *(h8*)(dst + 8) = hi;
        }
    };

    issue_stage(0, 0);

    for (int kc = 0; kc < KC; ++kc) {
        // ensure this wave's async writes (and ds_stores via barrier's dscnt
        // wait) are complete, then rendezvous: buffer kc&1 is ready for all.
        wait_asynccnt0();
        __syncthreads();

        if (kc + 1 < KC) issue_stage(kc + 1, (kc + 1) & 1);

        const _Float16* buf = &lda[kc & 1][0];

        // ---- A fragments for this wave's two 16-row tiles ----
        AFrag af[2];
        #pragma unroll
        for (int mi = 0; mi < 2; ++mi) {
            const int r  = wave_m * 32 + mi * 16 + ln16;
            const int kb = half * 8;
            af[mi].h[0] = *(const h8*)&buf[r * 40 + kb];
            af[mi].h[1] = *(const h8*)&buf[r * 40 + kb + 16];
        }

        // ---- 4 N-tiles x 2 M-tiles WMMA ----
        #pragma unroll
        for (int ni = 0; ni < 4; ++ni) {
            const int ntile = wave_n * 4 + ni;
            const int f     = ntile * KC + kc;
            AFrag bf;
            const h8* bp = (const h8*)(Bw + ((size_t)f * 32 + lane) * 16);
            bf.h[0] = bp[0];
            bf.h[1] = bp[1];
            #pragma unroll
            for (int mi = 0; mi < 2; ++mi) {
                acc[mi][ni] = __builtin_amdgcn_wmma_f32_16x16x32_f16(
                    false, af[mi].v, false, bf.v, (short)0, acc[mi][ni],
                    false, false);
            }
        }
    }

    // ---- epilogue: elementwise ops + fused LayerNorm ----
    float colg[4], colb[4], colbias[4];
    int   colidx[4];
    #pragma unroll
    for (int ni = 0; ni < 4; ++ni) {
        const int col = wave_n * 64 + ni * 16 + ln16;
        colidx[ni]  = col;
        colg[ni]    = ln_g[col];
        colb[ni]    = ln_b[col];
        colbias[ni] = bias ? bias[col] : 0.0f;
    }

    float rs[2][8], rs2[2][8];
    #pragma unroll
    for (int mi = 0; mi < 2; ++mi) {
        #pragma unroll
        for (int r = 0; r < 8; ++r) {
            const int row = blockRow + wave_m * 32 + mi * 16 + half * 8 + r;
            float scale = 1.0f;
            if (ROWSCALE) scale = (row < Mtot) ? rowscale[row] : 0.0f;
            float s = 0.0f, s2 = 0.0f;
            #pragma unroll
            for (int ni = 0; ni < 4; ++ni) {
                float c = acc[mi][ni][r];
                c += colbias[ni];
                if (RESID)
                    c += (row < Mtot)
                             ? resid[(size_t)row * GD + colidx[ni]]
                             : 0.0f;
                if (ROWSCALE) c *= scale;
                if (SILU) c = silu_f(c);
                acc[mi][ni][r] = c;
                s  += c;
                s2 += c * c;
            }
            // butterfly within the 16-lane C-fragment group (ds_swizzle imm)
            rs[mi][r]  = group16_sum(s);
            rs2[mi][r] = group16_sum(s2);
        }
    }

    if (ln16 == 0) {
        #pragma unroll
        for (int mi = 0; mi < 2; ++mi)
            #pragma unroll
            for (int r = 0; r < 8; ++r) {
                const int rit = wave_m * 32 + mi * 16 + half * 8 + r;
                redS[rit][wave_n]  = rs[mi][r];
                redS2[rit][wave_n] = rs2[mi][r];
            }
    }
    __syncthreads();

    #pragma unroll
    for (int mi = 0; mi < 2; ++mi) {
        #pragma unroll
        for (int r = 0; r < 8; ++r) {
            const int rit = wave_m * 32 + mi * 16 + half * 8 + r;
            const int row = blockRow + rit;
            const float S    = redS[rit][0] + redS[rit][1];
            const float S2   = redS2[rit][0] + redS2[rit][1];
            const float mean = S * (1.0f / 128.0f);
            const float var  = S2 * (1.0f / 128.0f) - mean * mean;
            const float rsig = rsqrtf(var + LN_EPS);
            if (row < Mtot) {
                #pragma unroll
                for (int ni = 0; ni < 4; ++ni) {
                    const float o =
                        (acc[mi][ni][r] - mean) * rsig * colg[ni] + colb[ni];
                    if (OUTF32)
                        ((float*)out)[(size_t)row * GD + colidx[ni]] = o;
                    else
                        ((_Float16*)out)[(size_t)row * GD + colidx[ni]] =
                            (_Float16)o;
                }
            }
        }
    }
}

// ---------------------------------------------------------------------------
// kq GEMM: [ROWS,128] x [128,32] (cols 0..15 = Wk, 16..31 = Wq), then
// w0[row] = mean_i(k_i * q_i). Block 256 = 8 waves, each wave 16 rows x 32 cols.
// A staged via async global->LDS, double buffered.
// ---------------------------------------------------------------------------
__global__ __launch_bounds__(256) void gemm_kq_kernel(
    const _Float16* __restrict__ xh, const _Float16* __restrict__ Bw,
    float* __restrict__ wsum, float* __restrict__ witer, int Mtot) {
    __shared__ __align__(16) _Float16 lda[2][128 * 40];
    const int tid  = threadIdx.x;
    const int wid  = tid >> 5;
    const int lane = tid & 31;
    const int half = lane >> 4;
    const int ln16 = lane & 15;
    const int blockRow = blockIdx.x * 128;

    const int srr = tid >> 1;
    const int skk = (tid & 1) * 16;
    int srow = blockRow + srr;
    if (srow > Mtot - 1) srow = Mtot - 1;

    v8f acc[2];
    acc[0] = zero_v8f();
    acc[1] = zero_v8f();

    auto issue_stage = [&](int kc, int bufIdx) {
        _Float16* dst = &lda[bufIdx][srr * 40 + skk];
        const _Float16* S = xh + (size_t)srow * GD + kc * 32 + skk;
        async_copy_32B(S, dst);
    };

    issue_stage(0, 0);

    for (int kc = 0; kc < 4; ++kc) {
        wait_asynccnt0();
        __syncthreads();
        if (kc + 1 < 4) issue_stage(kc + 1, (kc + 1) & 1);

        const _Float16* buf = &lda[kc & 1][0];
        AFrag af;
        {
            const int r  = wid * 16 + ln16;
            const int kb = half * 8;
            af.h[0] = *(const h8*)&buf[r * 40 + kb];
            af.h[1] = *(const h8*)&buf[r * 40 + kb + 16];
        }
        #pragma unroll
        for (int nt = 0; nt < 2; ++nt) {
            const int f = nt * 4 + kc;
            AFrag bf;
            const h8* bp = (const h8*)(Bw + ((size_t)f * 32 + lane) * 16);
            bf.h[0] = bp[0];
            bf.h[1] = bp[1];
            acc[nt] = __builtin_amdgcn_wmma_f32_16x16x32_f16(
                false, af.v, false, bf.v, (short)0, acc[nt], false, false);
        }
    }

    // w0 = (1/16) * sum_i k_i * q_i ; i striped across the 16-lane group
    #pragma unroll
    for (int r = 0; r < 8; ++r) {
        float p = group16_sum(acc[0][r] * acc[1][r]);
        if (ln16 == 0) {
            const int row = blockRow + wid * 16 + half * 8 + r;
            if (row < Mtot) {
                const float w0 = p * (1.0f / 16.0f);
                wsum[row]  = w0;
                witer[row] = w0;
            }
        }
    }
}

// ---------------------------------------------------------------------------
// Host-side launch
// ---------------------------------------------------------------------------
static inline size_t alignup(size_t x) { return (x + 255) & ~(size_t)255; }

extern "C" void kernel_launch(void* const* d_in, const int* in_sizes, int n_in,
                              void* d_out, int out_size, void* d_ws,
                              size_t ws_size, hipStream_t stream) {
    (void)in_sizes; (void)n_in; (void)out_size; (void)ws_size;
    const float* x         = (const float*)d_in[0];
    const int*   edge_idx  = (const int*)d_in[1];
    const float* edge_vals = (const float*)d_in[2];
    const float* sage_W    = (const float*)d_in[3];
    const float* sage_b    = (const float*)d_in[4];
    const float* sage_aggW = (const float*)d_in[5];
    const float* sage_ln_g = (const float*)d_in[6];
    const float* sage_ln_b = (const float*)d_in[7];
    const float* att_Wk    = (const float*)d_in[8];
    const float* att_Wq    = (const float*)d_in[9];
    const float* att_Wv    = (const float*)d_in[10];
    const float* att_ln_g  = (const float*)d_in[11];
    const float* att_ln_b  = (const float*)d_in[12];
    const float* lin1_W    = (const float*)d_in[13];
    const float* lin1_b    = (const float*)d_in[14];
    const float* lin2_W    = (const float*)d_in[15];
    const float* lin2_b    = (const float*)d_in[16];
    const float* ln1_g     = (const float*)d_in[17];
    const float* ln1_b     = (const float*)d_in[18];
    const float* ln2_g     = (const float*)d_in[19];
    const float* ln2_b     = (const float*)d_in[20];

    // ---- workspace carve ----
    char* p = (char*)d_ws;
    size_t off = 0;
    auto carve = [&](size_t bytes) {
        void* r = p + off;
        off = alignup(off + bytes);
        return r;
    };
    _Float16* xh       = (_Float16*)carve((size_t)ROWS * GD * 2);
    float*    agg      = (float*)   carve((size_t)ROWS * GD * 4);
    _Float16* sage_out = (_Float16*)carve((size_t)ROWS * GD * 2);
    _Float16* att_out  = (_Float16*)carve((size_t)ROWS * GD * 2);
    _Float16* h1       = (_Float16*)carve((size_t)ROWS * GD * 2);
    float*    wsum     = (float*)   carve((size_t)ROWS * 4);
    float*    wA       = (float*)   carve((size_t)ROWS * 4);
    float*    wB       = (float*)   carve((size_t)ROWS * 4);
    _Float16* whSage   = (_Float16*)carve((size_t)256 * 128 * 2);
    _Float16* whV      = (_Float16*)carve((size_t)128 * 128 * 2);
    _Float16* whL1     = (_Float16*)carve((size_t)256 * 128 * 2);
    _Float16* whL2     = (_Float16*)carve((size_t)128 * 128 * 2);
    _Float16* whKQ     = (_Float16*)carve((size_t)128 * 32 * 2);

    const int BIG = 1 << 30;
    const int MB  = (ROWS + 127) / 128;   // 1563 row-tiles

    // 1) x -> f16
    {
        const int n = ROWS * GD;
        cvt_f32_to_f16_kernel<<<(n + 255) / 256, 256, 0, stream>>>(x, xh, n);
    }
    // 2) weight swizzles
    swizzle_w_kernel<<<(256 * 128 + 255) / 256, 256, 0, stream>>>(
        sage_W, 128, sage_aggW, 128, /*split=*/128, /*by_k=*/1, 256, 128, whSage);
    swizzle_w_kernel<<<(128 * 128 + 255) / 256, 256, 0, stream>>>(
        att_Wv, 128, att_Wv, 128, BIG, 0, 128, 128, whV);
    swizzle_w_kernel<<<(256 * 128 + 255) / 256, 256, 0, stream>>>(
        lin1_W, 128, lin1_W, 128, BIG, 1, 256, 128, whL1);
    swizzle_w_kernel<<<(128 * 128 + 255) / 256, 256, 0, stream>>>(
        lin2_W, 128, lin2_W, 128, BIG, 0, 128, 128, whL2);
    swizzle_w_kernel<<<(128 * 32 + 255) / 256, 256, 0, stream>>>(
        att_Wk, 16, att_Wq, 16, /*split=*/16, /*by_k=*/0, 128, 32, whKQ);

    // 3) aggregation SpMM (zero + scatter-add)
    hipMemsetAsync(agg, 0, (size_t)ROWS * GD * 4, stream);
    spmm_agg_kernel<<<GE, 128, 0, stream>>>(edge_idx, edge_vals, x, agg, GE);

    // 4) retention weights: w0 = mean(k*q) via WMMA, then 2 propagation steps
    gemm_kq_kernel<<<MB, 256, 0, stream>>>(xh, whKQ, wsum, wA, ROWS);
    hipMemsetAsync(wB, 0, (size_t)ROWS * 4, stream);
    wprop_kernel<<<(GE + 255) / 256, 256, 0, stream>>>(edge_idx, edge_vals, wA,
                                                       wB, GE);
    waccum_kernel<<<(ROWS + 255) / 256, 256, 0, stream>>>(wsum, wB, ROWS);
    hipMemsetAsync(wA, 0, (size_t)ROWS * 4, stream);
    wprop_kernel<<<(GE + 255) / 256, 256, 0, stream>>>(edge_idx, edge_vals, wB,
                                                       wA, GE);
    waccum_kernel<<<(ROWS + 255) / 256, 256, 0, stream>>>(wsum, wA, ROWS);

    // 5) SAGE branch: LN(silu([x | agg] @ [sage_W; sage_aggW] + sage_b))
    gemm_ln_kernel<256, true, false, true, false, false, false>
        <<<MB, 256, 0, stream>>>(xh, agg, whSage, sage_b, nullptr, nullptr,
                                 sage_ln_g, sage_ln_b, sage_out, ROWS);

    // 6) Attention branch: LN((x @ Wv) * wsum[row])
    gemm_ln_kernel<128, true, true, false, true, false, false>
        <<<MB, 256, 0, stream>>>(xh, nullptr, whV, nullptr, wsum, nullptr,
                                 att_ln_g, att_ln_b, att_out, ROWS);

    // 7) Fusion 1: LN(silu([sage_out | att_out] @ lin1_W + lin1_b))
    gemm_ln_kernel<256, true, true, true, false, false, false>
        <<<MB, 256, 0, stream>>>(sage_out, att_out, whL1, lin1_b, nullptr,
                                 nullptr, ln1_g, ln1_b, h1, ROWS);

    // 8) Output: LN(silu(h1 @ lin2_W + lin2_b + x))   -> f32 d_out
    gemm_ln_kernel<128, true, true, true, false, true, true>
        <<<MB, 256, 0, stream>>>(h1, nullptr, whL2, lin2_b, nullptr, x, ln2_g,
                                 ln2_b, d_out, ROWS);
}
